// ScoreNet_670014898637
// MI455X (gfx1250) — matrix operand order repaired and checked by reading, not compiled, not run on hardware
//
#include <hip/hip_runtime.h>
#include <hip/hip_bf16.h>

typedef __attribute__((ext_vector_type(16))) _Float16 v16h;
typedef __attribute__((ext_vector_type(8)))  float    v8f;

#define kB    512
#define kN    22
#define kH    128
#define kEPG  462        // edges per graph (22*21), i-major contiguous
#define kTiles 29        // ceil(462/16)
#define kNW   8          // waves per block (256 threads)
#define kThreads 256
#define kSigma 0.168f

union Frag16 { uint4 u[2]; v16h h; _Float16 e[16]; };

// A-fragment (16x32 f16, M=edges, K=features) from row-major f16 LDS.
// Layout: lanes 0-15 hold row M=lane, K = k0..k0+15;
//         lanes 16-31 hold row M=lane-16, K = k0+16..k0+31.
static __device__ __forceinline__ v16h lds_a_frag(const _Float16* rowbase, int k0, int half8) {
  Frag16 t;
  const _Float16* p = rowbase + k0 + half8 * 16;
  t.u[0] = *reinterpret_cast<const uint4*>(p);
  t.u[1] = *reinterpret_cast<const uint4*>(p + 8);
  return t.h;
}
// B-fragment (32x16 f16, K x N): pre-swizzled in LDS so lane holds 16 contiguous halves.
// (lanes 0-15: col=lane, K=0..15; lanes 16-31: col=lane-16, K=16..31)
static __device__ __forceinline__ v16h lds_b_frag(const _Float16* fragbase, int lane) {
  Frag16 t;
  const _Float16* p = fragbase + lane * 16;
  t.u[0] = *reinterpret_cast<const uint4*>(p);
  t.u[1] = *reinterpret_cast<const uint4*>(p + 8);
  return t.h;
}
static __device__ __forceinline__ float siluf(float v) { return v / (1.f + __expf(-v)); }
// reduce across the 16 lanes of each half-wave (keeps lane bit4 fixed)
static __device__ __forceinline__ float half16_reduce(float v) {
  v += __shfl_xor(v, 1, 32);
  v += __shfl_xor(v, 2, 32);
  v += __shfl_xor(v, 4, 32);
  v += __shfl_xor(v, 8, 32);
  return v;
}

// ---------------- prep: x = xt*c_in ; h = [time_emb, atom_emb] @ W_in + b_in -------------
__global__ void __launch_bounds__(128)
prep_kernel(const float* __restrict__ xt, const float* __restrict__ t,
            const float* __restrict__ atom_table, const int* __restrict__ atom_types,
            const float* __restrict__ W_in, const float* __restrict__ b_in,
            float* __restrict__ x, float* __restrict__ h) {
  const int node = blockIdx.x;
  const int g = node / kN, a = node % kN;
  const int tid = threadIdx.x;
  __shared__ float feat[64];
  if (tid < 16) {
    float u  = logf(t[g]) * 0.25f;
    float fr = __expf(-logf(10000.f) * (float)tid / 16.f);
    float ang = u * fr;
    feat[tid]      = sinf(ang);
    feat[16 + tid] = cosf(ang);
  } else if (tid >= 32 && tid < 64) {
    int q = tid - 32;
    feat[32 + q] = atom_table[atom_types[a] * 32 + q];
  }
  if (tid < 3) {
    float tg = t[g];
    float cin = rsqrtf(tg * tg + kSigma * kSigma);
    x[node * 3 + tid] = xt[node * 3 + tid] * cin;
  }
  __syncthreads();
  const int c = tid;
  float acc = b_in[c];
  for (int k = 0; k < 64; ++k) acc += feat[k] * W_in[k * 128 + c];
  h[node * 128 + c] = acc;
}

// ---------------- dist_sq0 at input scale --------------------------------------------
__global__ void edge0_kernel(const float* __restrict__ x, float* __restrict__ d0) {
  int e = blockIdx.x * blockDim.x + threadIdx.x;
  if (e >= kB * kEPG) return;
  int g = e / kEPG, le = e % kEPG;
  int i = le / 21, jj = le - i * 21;
  int j = jj + (jj >= i ? 1 : 0);
  int bi = (g * kN + i) * 3, bj = (g * kN + j) * 3;
  float dx = x[bi] - x[bj], dy = x[bi + 1] - x[bj + 1], dz = x[bi + 2] - x[bj + 2];
  d0[e] = dx * dx + dy * dy + dz * dz;
}

// ---------------- fused edge kernel: one workgroup per graph -------------------------
__global__ void __launch_bounds__(kThreads)
egnn_edge_kernel(const float* __restrict__ x, const float* __restrict__ h,
                 const float* __restrict__ d0g, const float* __restrict__ bondg,
                 const float* __restrict__ ew1, const float* __restrict__ eb1,
                 const float* __restrict__ ew2, const float* __restrict__ eb2,
                 const float* __restrict__ aw, const float* __restrict__ ab,
                 const float* __restrict__ cw1, const float* __restrict__ cb1,
                 const float* __restrict__ cw2,
                 float* __restrict__ aggHg, float* __restrict__ aggXg, int l) {
  // weights pre-swizzled into B-fragment order: [kf][nf][lane][16 halves]
  __shared__ __align__(16) _Float16 sW1[9 * 8 * 512];  // 73.7 KB  (K padded 259->288)
  __shared__ __align__(16) _Float16 sW2[4 * 8 * 512];  // 32.8 KB
  __shared__ __align__(16) _Float16 sC1[4 * 8 * 512];  // 32.8 KB
  __shared__ __align__(16) _Float16 sH[kN * kH];       // h in f16, row-major
  __shared__ float sX[kN * 3];
  __shared__ float sEB1[128], sEB2[128], sCB1[128], sAW[128], sCW2[128];
  __shared__ float sD0[kEPG], sBM[kEPG];
  __shared__ float sAggH[kN * kH];
  __shared__ float sAggX[kN * 3];
  __shared__ __align__(16) float4 sStgRaw[kNW][256];   // 4 KB / wave f16 staging

  const int tid = threadIdx.x;
  const int lane = tid & 31;
  const int wave = tid >> 5;
  const int g = blockIdx.x;
  const int half8 = lane >> 4;
  const int col = lane & 15;

  // hint the L2 about this layer's weight slabs (global_prefetch_b8 path)
  if (tid == 0) {
    __builtin_prefetch(ew1 + l * 259 * 128, 0, 1);
    __builtin_prefetch(ew2 + l * 128 * 128, 0, 1);
    __builtin_prefetch(cw1 + l * 128 * 128, 0, 1);
  }

  // ---- cooperative loads / swizzles ----
  for (int fi = tid; fi < 9 * 8 * 512; fi += kThreads) {
    int frag = fi >> 9, r = fi & 511;
    int ln = r >> 4, q = r & 15;
    int kf = frag >> 3, nf = frag & 7;
    int k = kf * 32 + (ln >> 4) * 16 + q;
    int n = nf * 16 + (ln & 15);
    float v = (k < 259) ? ew1[(l * 259 + k) * 128 + n] : 0.f;
    sW1[fi] = (_Float16)v;
  }
  for (int fi = tid; fi < 4 * 8 * 512; fi += kThreads) {
    int frag = fi >> 9, r = fi & 511;
    int ln = r >> 4, q = r & 15;
    int kf = frag >> 3, nf = frag & 7;
    int k = kf * 32 + (ln >> 4) * 16 + q;
    int n = nf * 16 + (ln & 15);
    sW2[fi] = (_Float16)ew2[(l * 128 + k) * 128 + n];
    sC1[fi] = (_Float16)cw1[(l * 128 + k) * 128 + n];
  }
  if (tid < 128) {
    sEB1[tid] = eb1[l * 128 + tid]; sEB2[tid] = eb2[l * 128 + tid]; sCB1[tid] = cb1[l * 128 + tid];
    sAW[tid] = aw[l * 128 + tid];   sCW2[tid] = cw2[l * 128 + tid];
  }
  for (int idx = tid; idx < kN * kH; idx += kThreads) {
    sH[idx] = (_Float16)h[g * (kN * kH) + idx];
    sAggH[idx] = 0.f;
  }
  for (int idx = tid; idx < kN * 3; idx += kThreads) { sX[idx] = x[g * (kN * 3) + idx]; sAggX[idx] = 0.f; }
  for (int e = tid; e < kEPG; e += kThreads) { sD0[e] = d0g[g * kEPG + e]; sBM[e] = bondg[g * kEPG + e]; }
  const float abv = ab[l];
  __syncthreads();

  _Float16* stg16 = reinterpret_cast<_Float16*>(&sStgRaw[wave][0]);
  const int rbase = half8 * 8;

  // per-lane column-invariant weights/biases, hoisted out of the tile loop
  float eb1v[8], eb2v[8], cb1v[8], awv[8], cw2v[8];
  #pragma unroll
  for (int nf = 0; nf < 8; ++nf) {
    eb1v[nf] = sEB1[nf * 16 + col];
    eb2v[nf] = sEB2[nf * 16 + col];
    cb1v[nf] = sCB1[nf * 16 + col];
    awv[nf]  = sAW[nf * 16 + col];
    cw2v[nf] = sCW2[nf * 16 + col];
  }

  for (int tIdx = wave; tIdx < kTiles; tIdx += kNW) {
    const int eBase = tIdx * 16;
    // per-lane edge (row m = col)
    const int  eL = eBase + col;
    const bool validL = (eL < kEPG);
    const int  eC = validL ? eL : 0;
    const int  i_ = eC / 21;
    const int  jj = eC - i_ * 21;
    const int  j_ = jj + (jj >= i_ ? 1 : 0);
    const float dx0 = sX[i_ * 3 + 0] - sX[j_ * 3 + 0];
    const float dx1 = sX[i_ * 3 + 1] - sX[j_ * 3 + 1];
    const float dx2 = sX[i_ * 3 + 2] - sX[j_ * 3 + 2];
    const float radial = dx0 * dx0 + dx1 * dx1 + dx2 * dx2;
    const float inv = 1.f / (sqrtf(radial) + 1.f);   // NORM_CONST = 1.0
    const float cd0 = dx0 * inv, cd1 = dx1 * inv, cd2 = dx2 * inv;
    const float d0v = sD0[eC], bmv = sBM[eC];
    const _Float16* rowHi = sH + i_ * 128;
    const _Float16* rowHj = sH + j_ * 128;

    v8f acc[8];
    v16h bbuf0[8], bbuf1[8];   // double-buffered B fragments

    // ---- GEMM1: [16x288] @ W1 -> m1 ----
    #pragma unroll
    for (int nf = 0; nf < 8; ++nf)
      #pragma unroll
      for (int r = 0; r < 8; ++r) acc[nf][r] = 0.f;
    #pragma unroll
    for (int nf = 0; nf < 8; ++nf) bbuf0[nf] = lds_b_frag(sW1 + nf * 512, lane);
    #pragma unroll
    for (int kf = 0; kf < 9; ++kf) {
      v16h (&bc)[8] = (kf & 1) ? bbuf1 : bbuf0;
      v16h (&bn)[8] = (kf & 1) ? bbuf0 : bbuf1;
      v16h a;
      if (kf < 4)       a = lds_a_frag(rowHi, kf * 32, half8);
      else if (kf < 8)  a = lds_a_frag(rowHj, (kf - 4) * 32, half8);
      else {
        Frag16 af;
        #pragma unroll
        for (int q = 0; q < 16; ++q) af.e[q] = (_Float16)0.f;
        if (half8 == 0) { af.e[0] = (_Float16)radial; af.e[1] = (_Float16)d0v; af.e[2] = (_Float16)bmv; }
        a = af.h;
      }
      if (kf < 8) {
        #pragma unroll
        for (int nf = 0; nf < 8; ++nf) bn[nf] = lds_b_frag(sW1 + ((kf + 1) * 8 + nf) * 512, lane);
      }
      #pragma unroll
      for (int nf = 0; nf < 8; ++nf)
        acc[nf] = __builtin_amdgcn_wmma_f32_16x16x32_f16(false, a, false, bc[nf], (short)0, acc[nf], false, false);
    }
    // bias + silu, stage f16 (C layout -> row-major for next A loads)
    #pragma unroll
    for (int nf = 0; nf < 8; ++nf) {
      #pragma unroll
      for (int r = 0; r < 8; ++r) {
        float v = siluf(acc[nf][r] + eb1v[nf]);
        stg16[(rbase + r) * 128 + nf * 16 + col] = (_Float16)v;
      }
    }
    asm volatile("s_wait_dscnt 0" ::: "memory");

    // ---- GEMM2: m1 @ W2 -> m ----
    #pragma unroll
    for (int nf = 0; nf < 8; ++nf)
      #pragma unroll
      for (int r = 0; r < 8; ++r) acc[nf][r] = 0.f;
    #pragma unroll
    for (int nf = 0; nf < 8; ++nf) bbuf0[nf] = lds_b_frag(sW2 + nf * 512, lane);
    #pragma unroll
    for (int kf = 0; kf < 4; ++kf) {
      v16h (&bc)[8] = (kf & 1) ? bbuf1 : bbuf0;
      v16h (&bn)[8] = (kf & 1) ? bbuf0 : bbuf1;
      v16h a = lds_a_frag(stg16 + col * 128, kf * 32, half8);
      if (kf < 3) {
        #pragma unroll
        for (int nf = 0; nf < 8; ++nf) bn[nf] = lds_b_frag(sW2 + ((kf + 1) * 8 + nf) * 512, lane);
      }
      #pragma unroll
      for (int nf = 0; nf < 8; ++nf)
        acc[nf] = __builtin_amdgcn_wmma_f32_16x16x32_f16(false, a, false, bc[nf], (short)0, acc[nf], false, false);
    }
    #pragma unroll
    for (int nf = 0; nf < 8; ++nf)
      #pragma unroll
      for (int r = 0; r < 8; ++r)
        acc[nf][r] = siluf(acc[nf][r] + eb2v[nf]);

    // ---- attention gate: sigmoid(m . aw + ab) via register dot + lane butterfly ----
    float gate[8];
    #pragma unroll
    for (int r = 0; r < 8; ++r) {
      float p = 0.f;
      #pragma unroll
      for (int nf = 0; nf < 8; ++nf) p += acc[nf][r] * awv[nf];
      p = half16_reduce(p);                       // all lanes of this half hold row rbase+r's dot
      const int row = rbase + r;
      const bool vrow = (eBase + row) < kEPG;
      gate[r] = vrow ? (1.f / (1.f + __expf(-(p + abv)))) : 0.f;
    }

    // gated stage (for GEMM3 A) + segment-sum of m into agg_h (LDS float atomics)
    #pragma unroll
    for (int r = 0; r < 8; ++r) {
      const int row = rbase + r;
      int er = eBase + row; if (er >= kEPG) er = 0;
      const int nodeR = er / 21;
      #pragma unroll
      for (int nf = 0; nf < 8; ++nf) {
        float mg = acc[nf][r] * gate[r];
        stg16[row * 128 + nf * 16 + col] = (_Float16)mg;
        atomicAdd(&sAggH[nodeR * 128 + nf * 16 + col], mg);
      }
    }
    asm volatile("s_wait_dscnt 0" ::: "memory");

    // ---- GEMM3: m_gated @ cw1 -> p ----
    #pragma unroll
    for (int nf = 0; nf < 8; ++nf)
      #pragma unroll
      for (int r = 0; r < 8; ++r) acc[nf][r] = 0.f;
    #pragma unroll
    for (int nf = 0; nf < 8; ++nf) bbuf0[nf] = lds_b_frag(sC1 + nf * 512, lane);
    #pragma unroll
    for (int kf = 0; kf < 4; ++kf) {
      v16h (&bc)[8] = (kf & 1) ? bbuf1 : bbuf0;
      v16h (&bn)[8] = (kf & 1) ? bbuf0 : bbuf1;
      v16h a = lds_a_frag(stg16 + col * 128, kf * 32, half8);
      if (kf < 3) {
        #pragma unroll
        for (int nf = 0; nf < 8; ++nf) bn[nf] = lds_b_frag(sC1 + ((kf + 1) * 8 + nf) * 512, lane);
      }
      #pragma unroll
      for (int nf = 0; nf < 8; ++nf)
        acc[nf] = __builtin_amdgcn_wmma_f32_16x16x32_f16(false, a, false, bc[nf], (short)0, acc[nf], false, false);
    }
    // c = silu(p + cb1) . cw2 via register dot + lane butterfly; then trans into agg_x
    #pragma unroll
    for (int r = 0; r < 8; ++r) {
      float p = 0.f;
      #pragma unroll
      for (int nf = 0; nf < 8; ++nf) p += siluf(acc[nf][r] + cb1v[nf]) * cw2v[nf];
      p = half16_reduce(p);                       // row rbase+r's coordinate scalar
      // the lane whose own edge is this row does the coordinate update
      if (col == rbase + r && validL) {
        atomicAdd(&sAggX[i_ * 3 + 0], cd0 * p);
        atomicAdd(&sAggX[i_ * 3 + 1], cd1 * p);
        atomicAdd(&sAggX[i_ * 3 + 2], cd2 * p);
      }
    }
    asm volatile("s_wait_dscnt 0" ::: "memory");
  }

  __syncthreads();
  for (int idx = tid; idx < kN * kH; idx += kThreads) aggHg[g * (kN * kH) + idx] = sAggH[idx];
  for (int idx = tid; idx < kN * 3; idx += kThreads)  aggXg[g * (kN * 3) + idx]  = sAggX[idx];
}

// ---------------- node update: x += agg_x ; h += MLP([h, agg]) -----------------------
__global__ void __launch_bounds__(128)
node_kernel(float* __restrict__ x, float* __restrict__ h,
            const float* __restrict__ aggH, const float* __restrict__ aggX,
            const float* __restrict__ nw1, const float* __restrict__ nb1,
            const float* __restrict__ nw2, const float* __restrict__ nb2, int l) {
  const int node = blockIdx.x;
  const int c = threadIdx.x;
  __shared__ float hl[128], al[128], ul[128];
  hl[c] = h[node * 128 + c];
  al[c] = aggH[node * 128 + c];
  __syncthreads();
  float u = nb1[l * 128 + c];
  for (int k = 0; k < 128; ++k) u += hl[k] * nw1[(l * 256 + k) * 128 + c];
  for (int k = 0; k < 128; ++k) u += al[k] * nw1[(l * 256 + 128 + k) * 128 + c];
  ul[c] = siluf(u);
  __syncthreads();
  float o = nb2[l * 128 + c];
  for (int k = 0; k < 128; ++k) o += ul[k] * nw2[(l * 128 + k) * 128 + c];
  h[node * 128 + c] = hl[c] + o;
  if (c < 3) x[node * 3 + c] += aggX[node * 3 + c];
}

// ---------------- final: remove mean, EDM skip/out preconditioning -------------------
__global__ void __launch_bounds__(128)
final_kernel(const float* __restrict__ xt, const float* __restrict__ t,
             const float* __restrict__ x, float* __restrict__ out) {
  const int g = blockIdx.x;
  const int tid = threadIdx.x;
  __shared__ float mean[3];
  if (tid < 3) {
    float s = 0.f;
    for (int n = 0; n < kN; ++n) s += x[(g * kN + n) * 3 + tid];
    mean[tid] = s / (float)kN;
  }
  __syncthreads();
  if (tid < kN * 3) {
    int d = tid % 3;
    int idx = g * (kN * 3) + tid;
    float tg = t[g];
    float den = tg * tg + kSigma * kSigma;
    float cs = kSigma * kSigma / den;
    float co = tg * kSigma * rsqrtf(den);
    out[idx] = xt[idx] * cs + (x[idx] - mean[d]) * co;
  }
}

extern "C" void kernel_launch(void* const* d_in, const int* in_sizes, int n_in,
                              void* d_out, int out_size, void* d_ws, size_t ws_size,
                              hipStream_t stream) {
  const float* xt         = (const float*)d_in[0];
  const float* t          = (const float*)d_in[1];
  const float* atom_table = (const float*)d_in[2];
  const float* bond       = (const float*)d_in[3];
  const float* W_in       = (const float*)d_in[4];
  const float* b_in       = (const float*)d_in[5];
  const float* ew1        = (const float*)d_in[6];
  const float* eb1        = (const float*)d_in[7];
  const float* ew2        = (const float*)d_in[8];
  const float* eb2        = (const float*)d_in[9];
  const float* aw         = (const float*)d_in[10];
  const float* ab         = (const float*)d_in[11];
  const float* cw1        = (const float*)d_in[12];
  const float* cb1        = (const float*)d_in[13];
  const float* cw2        = (const float*)d_in[14];
  const float* nw1        = (const float*)d_in[15];
  const float* nb1        = (const float*)d_in[16];
  const float* nw2        = (const float*)d_in[17];
  const float* nb2        = (const float*)d_in[18];
  const int*   atom_types = (const int*)d_in[21];
  float* out = (float*)d_out;

  const int BN = kB * kN;                 // 11264
  float* ws   = (float*)d_ws;
  float* x    = ws;                       // BN*3      = 33792
  float* hbuf = x + BN * 3;               // BN*128    = 1441792
  float* agH  = hbuf + BN * 128;          // BN*128
  float* agX  = agH + BN * 128;           // BN*3
  float* d0   = agX + BN * 3;             // B*EPG     = 236544

  prep_kernel<<<BN, 128, 0, stream>>>(xt, t, atom_table, atom_types, W_in, b_in, x, hbuf);
  edge0_kernel<<<(kB * kEPG + 255) / 256, 256, 0, stream>>>(x, d0);
  for (int l = 0; l < 4; ++l) {
    egnn_edge_kernel<<<kB, kThreads, 0, stream>>>(x, hbuf, d0, bond,
                                                  ew1, eb1, ew2, eb2, aw, ab, cw1, cb1, cw2,
                                                  agH, agX, l);
    node_kernel<<<BN, 128, 0, stream>>>(x, hbuf, agH, agX, nw1, nb1, nw2, nb2, l);
  }
  final_kernel<<<kB, 128, 0, stream>>>(xt, t, x, out);
}
